// InteractionModule_31233002176786
// MI455X (gfx1250) — compile-verified, matrix operand order checked by reading, not run on hardware
//
#include <hip/hip_runtime.h>
#include <hip/hip_bf16.h>
#include <math.h>

typedef __attribute__((ext_vector_type(16))) _Float16 v16h;
typedef __attribute__((ext_vector_type(8)))  _Float16 v8h;
typedef __attribute__((ext_vector_type(8)))  float    v8f;

#define NN 10000                      // N_NODES (reference constant)
#define INV_SQRT3f 0.57735026918962576451f

// ---- B-fragment table (pre-swizzled f16 WMMA operands in workspace) ------
// frag f: 32 lanes x 16 halfs = 1KB. Index = base + ks*NT + nt.
#define FR_W1   0      // 5*8
#define FR_W2   40     // 4*8
#define FR_WENV 72     // 4*5
#define FR_WQ   92     // 5*16
#define FR_WD1  172    // 8
#define FR_WD2  180    // 8
#define FR_WK   188    // 16
#define FR_WS   204
#define FR_WV   205
#define FR_WLS  206
#define FR_WLV  207
#define FR_TOTAL 208

// ---------------------------------------------------------------- helpers
__device__ __forceinline__ v8f wmma16(v16h a, v16h b, v8f c) {
  // v_wmma_f32_16x16x32_f16 (probe-verified signature, 8 args)
  return __builtin_amdgcn_wmma_f32_16x16x32_f16(false, a, false, b, (short)0, c, false, false);
}

__device__ __forceinline__ v8f splat8(float v) {
  v8f r;
#pragma unroll
  for (int j = 0; j < 8; ++j) r[j] = v;
  return r;
}

__device__ __forceinline__ float hred16(float x) {
  x += __shfl_xor(x, 1, 32);
  x += __shfl_xor(x, 2, 32);
  x += __shfl_xor(x, 4, 32);
  x += __shfl_xor(x, 8, 32);
  return x;
}

// order-preserving float <-> uint encoding for atomicMax-based segment max
__device__ __forceinline__ unsigned fenc(float f) {
  unsigned b = __float_as_uint(f);
  return (b & 0x80000000u) ? ~b : (b | 0x80000000u);
}
__device__ __forceinline__ float fdec(unsigned u) {
  unsigned b = (u & 0x80000000u) ? (u & 0x7fffffffu) : ~u;
  return __uint_as_float(b);
}

// A fragment from LDS row-major [16][ld] f16. ISA 16-bit A layout is two
// contiguous 16B runs per lane -> two ds_load_b128.
__device__ __forceinline__ v16h a_frag(const _Float16* A, int ld, int k0) {
  int lane = threadIdx.x & 31;
  int r = lane & 15, hi = lane >> 4;
  const _Float16* p = A + r * ld + k0 + hi * 8;
  v8h lo = *(const v8h*)(p);
  v8h hh = *(const v8h*)(p + 16);
  v16h a;
#pragma unroll
  for (int j = 0; j < 8; ++j) { a[j] = lo[j]; a[8 + j] = hh[j]; }
  return a;
}

// B fragment: one 32B per-lane load from the pre-swizzled fragment table
__device__ __forceinline__ v16h bfrag_ld(const _Float16* __restrict__ FB, int f) {
  int lane = threadIdx.x & 31;
  return *(const v16h*)(FB + ((long)f * 32 + lane) * 16);
}

// ------------------------------------------------------- weight repack
// One wave per fragment: swizzle f32 weights into the wave32 WMMA B layout.
__global__ void repack_kernel(const float* __restrict__ W1, const float* __restrict__ W2,
                              const float* __restrict__ Wenv, const float* __restrict__ Wq,
                              const float* __restrict__ Wd1, const float* __restrict__ Wd2,
                              const float* __restrict__ Wk, const float* __restrict__ Ws,
                              const float* __restrict__ Wv, const float* __restrict__ Wls,
                              const float* __restrict__ Wlv, _Float16* __restrict__ FB)
{
  int f = blockIdx.x;
  int lane = threadIdx.x & 31;
  const float* W; int ldn, k0, n0, kmax;
  if (f < FR_W2)        { W = W1;  ldn = 128; kmax = 160; int g = f;           k0 = (g >> 3) * 32; n0 = (g & 7) * 16; }
  else if (f < FR_WENV) { W = W2;  ldn = 128; kmax = 128; int g = f - FR_W2;   k0 = (g >> 3) * 32; n0 = (g & 7) * 16; }
  else if (f < FR_WQ)   { W = Wenv; ldn = 80; kmax = 128; int g = f - FR_WENV; k0 = (g / 5) * 32;  n0 = (g % 5) * 16; }
  else if (f < FR_WD1)  { W = Wq;  ldn = 256; kmax = 160; int g = f - FR_WQ;   k0 = (g >> 4) * 32; n0 = (g & 15) * 16; }
  else if (f < FR_WD2)  { W = Wd1; ldn = 16;  kmax = 256; k0 = (f - FR_WD1) * 32; n0 = 0; }
  else if (f < FR_WK)   { W = Wd2; ldn = 16;  kmax = 256; k0 = (f - FR_WD2) * 32; n0 = 0; }
  else if (f < FR_WS)   { W = Wk;  ldn = 256; kmax = 16;  k0 = 0; n0 = (f - FR_WK) * 16; }
  else if (f == FR_WS)  { W = Ws;  ldn = 16;  kmax = 16;  k0 = 0; n0 = 0; }
  else if (f == FR_WV)  { W = Wv;  ldn = 16;  kmax = 16;  k0 = 0; n0 = 0; }
  else if (f == FR_WLS) { W = Wls; ldn = 16;  kmax = 32;  k0 = 0; n0 = 0; }
  else                  { W = Wlv; ldn = 16;  kmax = 32;  k0 = 0; n0 = 0; }

  int n = (lane & 15) + n0, hi = lane >> 4;
  v16h b;
#pragma unroll
  for (int j = 0; j < 16; ++j) {
    int k = k0 + hi * 8 + (j < 8 ? j : j + 8);
    b[j] = (k < kmax) ? (_Float16)W[(long)k * ldn + n] : (_Float16)0.0f;
  }
  *(v16h*)(FB + ((long)f * 32 + lane) * 16) = b;
}

// ---------------------------------------------------------------- kernel A
// One wave = one 16-edge tile; 2 waves / 64-thread block (LDS budget).
__global__ __launch_bounds__(64) void edge_main_kernel(
    const float* __restrict__ X, const float* __restrict__ eqfeat,
    const float* __restrict__ cutoff, const float* __restrict__ eattr,
    const float* __restrict__ ninv, const float* __restrict__ einv,
    const float* __restrict__ b1, const float* __restrict__ b2,
    const float* __restrict__ benv,
    const float* __restrict__ lng, const float* __restrict__ lnb,
    const float* __restrict__ lnqg, const float* __restrict__ lnqb,
    const float* __restrict__ bq,
    const float* __restrict__ lnkg, const float* __restrict__ lnkb,
    const float* __restrict__ bk,
    const int* __restrict__ active, const int* __restrict__ ecen,
    const int* __restrict__ eneigh,
    const _Float16* __restrict__ FB,
    float* __restrict__ out_lat, float* __restrict__ ws64,
    float* __restrict__ waBuf, unsigned int* __restrict__ mbuf, int E)
{
  __shared__ _Float16 Abuf[2][16 * 256];   // per-wave A staging (8KB)
  __shared__ _Float16 Wn[2][16 * 64];      // per-wave normalized env-weights
  __shared__ float    Xs[2][16 * 160];     // per-wave async f32 staging (10KB)

  const int wave = threadIdx.x >> 5;
  const int lane = threadIdx.x & 31;
  const int n = lane & 15, hi = lane >> 4;
  const long tile = (long)blockIdx.x * 2 + wave;
  const long e0 = tile * 16;
  if (e0 >= E) return;                     // wave-uniform
  _Float16* A = Abuf[wave];
  _Float16* WnT = Wn[wave];
  float* xs = Xs[wave];

  __builtin_prefetch(X + (e0 + 16) * 160, 0, 1);  // global_prefetch next tile

  // ---- Phase 1: async-copy X tile global->LDS (TDM-class path, ASYNCcnt),
  //               then convert to f16 A tile; h = silu(X @ W1 + b1) --------
  {
    unsigned ldsb = (unsigned)(unsigned long long)(void*)xs;
    for (int it = 0; it < 20; ++it) {
      int c = it * 32 + lane;              // 640 x 16B chunks (16 rows x 640B)
      int r = c / 40;
      int co = c % 40;
      long e = e0 + r; if (e >= E) e = E - 1;
      unsigned dst = ldsb + (unsigned)(r * 640 + co * 16);
      unsigned long long src = (unsigned long long)(const void*)(X + e * 160 + co * 4);
      asm volatile("global_load_async_to_lds_b128 %0, %1, off"
                   :: "v"(dst), "v"(src) : "memory");
    }
    asm volatile("s_wait_asynccnt 0" ::: "memory");
  }
  for (int r = 0; r < 16; ++r)
    for (int c = lane; c < 160; c += 32)
      A[r * 256 + c] = (_Float16)xs[r * 160 + c];

  v8f hacc[8];
#pragma unroll
  for (int nt = 0; nt < 8; ++nt) hacc[nt] = splat8(b1[nt * 16 + n]);
  for (int ks = 0; ks < 5; ++ks) {
    v16h af = a_frag(A, 256, ks * 32);
#pragma unroll
    for (int nt = 0; nt < 8; ++nt)
      hacc[nt] = wmma16(af, bfrag_ld(FB, FR_W1 + ks * 8 + nt), hacc[nt]);
  }
#pragma unroll
  for (int nt = 0; nt < 8; ++nt)
#pragma unroll
    for (int j = 0; j < 8; ++j) {
      float x = hacc[nt][j];
      x = x / (1.0f + __expf(-x));                 // silu
      A[(j + hi * 8) * 256 + nt * 16 + n] = (_Float16)x;
    }

  // ---- Phase 2: lat = (h @ W2 + b2) * cutoff ; scatter to latents out ----
  v8f lacc[8];
#pragma unroll
  for (int nt = 0; nt < 8; ++nt) lacc[nt] = splat8(b2[nt * 16 + n]);
  for (int ks = 0; ks < 4; ++ks) {
    v16h af = a_frag(A, 256, ks * 32);
#pragma unroll
    for (int nt = 0; nt < 8; ++nt)
      lacc[nt] = wmma16(af, bfrag_ld(FB, FR_W2 + ks * 8 + nt), lacc[nt]);
  }
#pragma unroll
  for (int j = 0; j < 8; ++j) {
    long e = e0 + j + hi * 8;
    bool ok = e < E;
    float cv = ok ? cutoff[e] : 0.0f;
    long ae = ok ? (long)active[e] : 0;
#pragma unroll
    for (int nt = 0; nt < 8; ++nt) {
      float x = lacc[nt][j] * cv;
      if (ok) out_lat[ae * 128 + nt * 16 + n] = x;
      A[(j + hi * 8) * 256 + nt * 16 + n] = (_Float16)x;
    }
  }

  // ---- Phase 3: weights = LN80(lat @ Wenv + benv) ------------------------
  v8f wacc[5];
#pragma unroll
  for (int nt = 0; nt < 5; ++nt) wacc[nt] = splat8(benv[nt * 16 + n]);
  for (int ks = 0; ks < 4; ++ks) {
    v16h af = a_frag(A, 256, ks * 32);
#pragma unroll
    for (int nt = 0; nt < 5; ++nt)
      wacc[nt] = wmma16(af, bfrag_ld(FB, FR_WENV + ks * 5 + nt), wacc[nt]);
  }
#pragma unroll
  for (int j = 0; j < 8; ++j) {
    float s = 0.f, s2 = 0.f;
#pragma unroll
    for (int t = 0; t < 5; ++t) { float v = wacc[t][j]; s += v; s2 += v * v; }
    s = hred16(s); s2 = hred16(s2);
    float mu = s * (1.0f / 80.0f);
    float var = s2 * (1.0f / 80.0f) - mu * mu;
    float rs = rsqrtf(var + 1e-5f);
#pragma unroll
    for (int t = 0; t < 5; ++t) wacc[t][j] = (wacc[t][j] - mu) * rs;
  }
#pragma unroll
  for (int t = 0; t < 5; ++t) {
    float g = lng[t * 16 + n], bb = lnb[t * 16 + n];
#pragma unroll
    for (int j = 0; j < 8; ++j) {
      float w = wacc[t][j] * g + bb;
      if (t < 4) {                                // only first 64 cols used
        int row = j + hi * 8;
        WnT[row * 64 + t * 16 + n] = (_Float16)w;
        long e = e0 + row;
        if (e < E) ws64[e * 64 + t * 16 + n] = w;
      }
    }
  }

  // ---- Phase 4: Q = LN160(efa) @ Wq + bq (2 lanes per edge for LN) -------
  {
    int r = lane >> 1, hh = lane & 1;
    long e = e0 + r; if (e >= E) e = E - 1;
    long ec = ecen[e], en = eneigh[e];
    float s = 0.f, s2 = 0.f;
    for (int k = 0; k < 80; ++k) {
      int col = hh * 80 + k;
      float v = (col < 64)  ? ninv[ec * 64 + col]
              : (col < 128) ? ninv[en * 64 + col - 64]
                            : einv[e * 32 + col - 128];
      s += v; s2 += v * v;
    }
    s  += __shfl_xor(s, 1, 32);
    s2 += __shfl_xor(s2, 1, 32);
    float mu = s * (1.0f / 160.0f);
    float var = s2 * (1.0f / 160.0f) - mu * mu;
    float rs = rsqrtf(var + 1e-5f);
    for (int k = 0; k < 80; ++k) {
      int col = hh * 80 + k;
      float v = (col < 64)  ? ninv[ec * 64 + col]
              : (col < 128) ? ninv[en * 64 + col - 64]
                            : einv[e * 32 + col - 128];
      A[r * 256 + col] = (_Float16)((v - mu) * rs * lnqg[col] + lnqb[col]);
    }
  }
  v8f qacc[16];
#pragma unroll
  for (int t = 0; t < 16; ++t) qacc[t] = splat8(bq[t * 16 + n]);
  for (int ks = 0; ks < 5; ++ks) {
    v16h af = a_frag(A, 256, ks * 32);
#pragma unroll
    for (int t = 0; t < 16; ++t)
      qacc[t] = wmma16(af, bfrag_ld(FB, FR_WQ + ks * 16 + t), qacc[t]);
  }

  // ---- Phase 5: dot = outer(s1,s2)@Wd1 + outer-v@Wd2 (rank-1 factored) ---
  v8f dacc = splat8(0.0f);
  for (int r = 0; r < 16; ++r) {
    long e = e0 + r; if (e >= E) e = E - 1;
    float alpha = eattr[e * 4 + 0] * eqfeat[e * 4 + 0];
#pragma unroll
    for (int kk = 0; kk < 8; ++kk) {
      int k = lane * 8 + kk;
      int u = k >> 4, v = k & 15;
      float au = (float)WnT[r * 64 + 32 + 2 * u];   // emb scalar weights
      float bv = (float)WnT[r * 64 + 2 * v];        // eqf scalar weights
      A[r * 256 + k] = (_Float16)(alpha * au * bv);
    }
  }
  for (int ks = 0; ks < 8; ++ks)
    dacc = wmma16(a_frag(A, 256, ks * 32), bfrag_ld(FB, FR_WD1 + ks), dacc);

  for (int r = 0; r < 16; ++r) {
    long e = e0 + r; if (e >= E) e = E - 1;
    float beta = INV_SQRT3f * (eattr[e * 4 + 1] * eqfeat[e * 4 + 1] +
                               eattr[e * 4 + 2] * eqfeat[e * 4 + 2] +
                               eattr[e * 4 + 3] * eqfeat[e * 4 + 3]);
#pragma unroll
    for (int kk = 0; kk < 8; ++kk) {
      int k = lane * 8 + kk;
      int u = k >> 4, v = k & 15;
      float cu = (float)WnT[r * 64 + 33 + 2 * u];   // emb vector weights
      float dv = (float)WnT[r * 64 + 2 * v + 1];    // eqf vector weights
      A[r * 256 + k] = (_Float16)(beta * cu * dv);
    }
  }
  for (int ks = 0; ks < 8; ++ks)
    dacc = wmma16(a_frag(A, 256, ks * 32), bfrag_ld(FB, FR_WD2 + ks), dacc);

  // ---- Phase 6: Y = LN16(dot), staged padded to K=32 ---------------------
#pragma unroll
  for (int j = 0; j < 8; ++j) {
    float x = dacc[j];
    float s = hred16(x);
    float s2 = hred16(x * x);
    float mu = s * (1.0f / 16.0f);
    float var = s2 * (1.0f / 16.0f) - mu * mu;
    float y = (x - mu) * rsqrtf(var + 1e-5f) * lnkg[n] + lnkb[n];
    int row = j + hi * 8;
    A[row * 256 + n] = (_Float16)y;
    A[row * 256 + 16 + n] = (_Float16)0.0f;
  }

  // ---- Phase 7: per head t, K_t = Y @ Wk_t + bk ; Wa = 4 * sum_d Q*K -----
  v16h aY = a_frag(A, 256, 0);
  for (int t = 0; t < 16; ++t) {
    v8f kacc = splat8(bk[t * 16 + n]);
    kacc = wmma16(aY, bfrag_ld(FB, FR_WK + t), kacc);
#pragma unroll
    for (int j = 0; j < 8; ++j) {
      float p = hred16(qacc[t][j] * kacc[j]);
      if (n == t) {
        long e = e0 + j + hi * 8;
        if (e < E) {
          float wa = p * 4.0f;                      // ISQRTD
          waBuf[e * 16 + t] = wa;
          atomicMax(&mbuf[(long)ecen[e] * 16 + t], fenc(wa));
        }
      }
    }
  }
}

// ---------------------------------------------------------------- kernel B/C
__global__ void init_ws_kernel(unsigned* mbuf, float* den, float* env, int nm, int ne) {
  int i = blockIdx.x * blockDim.x + threadIdx.x;
  if (i < nm) { mbuf[i] = 0u; den[i] = 0.0f; }
  if (i < ne) env[i] = 0.0f;
}

__global__ void softmax_kernel(float* waEx, const int* __restrict__ ecen,
                               const unsigned* __restrict__ mbuf, float* den, int n16) {
  int i = blockIdx.x * blockDim.x + threadIdx.x;
  if (i >= n16) return;
  int e = i >> 4, t = i & 15;
  int nd = ecen[e];
  float m = fdec(mbuf[nd * 16 + t]);
  if (!(fabsf(m) < 3.0e38f)) m = 0.0f;             // non-finite -> 0 (ref)
  float ex = __expf(waEx[i] - m);
  waEx[i] = ex;
  atomicAdd(&den[nd * 16 + t], ex);
}

__global__ void env_accum_kernel(const float* __restrict__ waEx, const float* __restrict__ den,
                                 const float* __restrict__ ws64, const float* __restrict__ eattr,
                                 const int* __restrict__ ecen, float* env, int n16) {
  int i = blockIdx.x * blockDim.x + threadIdx.x;
  if (i >= n16) return;
  int e = i >> 4, t = i & 15;
  int nd = ecen[e];
  float attn = waEx[i] / fmaxf(den[nd * 16 + t], 1e-30f);
  float w0 = ws64[(long)e * 64 + 32 + 2 * t];
  float w1 = ws64[(long)e * 64 + 33 + 2 * t];
  const float* ea = eattr + (long)e * 4;
  atomicAdd(&env[(long)nd * 64 + t * 4 + 0], ea[0] * w0 * attn);
  atomicAdd(&env[(long)nd * 64 + t * 4 + 1], ea[1] * w1 * attn);
  atomicAdd(&env[(long)nd * 64 + t * 4 + 2], ea[2] * w1 * attn);
  atomicAdd(&env[(long)nd * 64 + t * 4 + 3], ea[3] * w1 * attn);
}

// ---------------------------------------------------------------- kernel D
__global__ __launch_bounds__(128) void env_xform_kernel(const float* __restrict__ env,
                                                        const _Float16* __restrict__ FB,
                                                        float* __restrict__ envT, int NNODES) {
  __shared__ _Float16 Abuf[4][16 * 32];
  int wave = threadIdx.x >> 5, lane = threadIdx.x & 31;
  int n = lane & 15, hi = lane >> 4;
  long tile = (long)blockIdx.x * 4 + wave;
  long n0 = tile * 16;
  if (n0 >= NNODES) return;
  _Float16* A = Abuf[wave];
  for (int c = 0; c < 4; ++c) {
    for (int r = 0; r < 16; ++r) {
      long nd = n0 + r; if (nd >= NNODES) nd = NNODES - 1;
      int col = lane;
      float v = (col < 16) ? env[nd * 64 + col * 4 + c] : 0.0f;
      A[r * 32 + col] = (_Float16)v;
    }
    v8f acc = splat8(0.0f);
    acc = wmma16(a_frag(A, 32, 0), bfrag_ld(FB, c == 0 ? FR_WS : FR_WV), acc);
#pragma unroll
    for (int j = 0; j < 8; ++j) {
      long nd = n0 + j + hi * 8;
      if (nd < NNODES) envT[nd * 64 + n * 4 + c] = acc[j];
    }
  }
}

// ---------------------------------------------------------------- kernel E
__global__ __launch_bounds__(128) void edge_out_kernel(
    const float* __restrict__ eqfeat, const float* __restrict__ ws64,
    const float* __restrict__ envT, const _Float16* __restrict__ FB,
    const int* __restrict__ active, const int* __restrict__ ecen,
    const float* __restrict__ out_lat,
    float* __restrict__ out_inv, float* __restrict__ out_eq, int E)
{
  __shared__ _Float16 Abuf[4][16 * 32];
  int wave = threadIdx.x >> 5, lane = threadIdx.x & 31;
  int n = lane & 15, hi = lane >> 4;
  long tile = (long)blockIdx.x * 4 + wave;
  long e0 = tile * 16;
  if (e0 >= E) return;
  _Float16* A = Abuf[wave];

  // inv_latent[:, :128] = lat_act gather
  for (int r = 0; r < 16; ++r) {
    long e = e0 + r; if (e >= E) break;
    long ae = active[e];
    for (int c = lane; c < 128; c += 32)
      out_inv[e * 160 + c] = out_lat[ae * 128 + c];
  }

  // scalar channels tp[:,:,0:2] -> A_s ; also inv_latent[:,128:160]
  for (int r = 0; r < 16; ++r) {
    long e = e0 + r; bool ok = e < E; long ee = ok ? e : E - 1;
    long nd = ecen[ee];
    int col = lane;                 // 0..31 = 2*m + jj
    int m = col >> 1, jj = col & 1;
    float w0 = ws64[ee * 64 + 2 * m], w1 = ws64[ee * 64 + 2 * m + 1];
    const float* ef = eqfeat + ee * 4;
    float as = ef[0] * w0;
    float bs = envT[nd * 64 + m * 4 + 0];
    float val;
    if (jj == 0) val = as * bs;
    else {
      float d = ef[1] * w1 * envT[nd * 64 + m * 4 + 1] +
                ef[2] * w1 * envT[nd * 64 + m * 4 + 2] +
                ef[3] * w1 * envT[nd * 64 + m * 4 + 3];
      val = INV_SQRT3f * d;
    }
    A[r * 32 + col] = (_Float16)val;
    if (ok) out_inv[e * 160 + 128 + col] = val;
  }
  {
    v8f sacc = splat8(0.0f);
    sacc = wmma16(a_frag(A, 32, 0), bfrag_ld(FB, FR_WLS), sacc);
#pragma unroll
    for (int j = 0; j < 8; ++j) {
      long e = e0 + j + hi * 8;
      if (e < E) out_eq[e * 64 + n * 4 + 0] = sacc[j];
    }
  }
  // vector channels: tp[:,:,2:].reshape(E,32,3)[..,i] -> A_v ; @ Wlv
  for (int i = 0; i < 3; ++i) {
    for (int r = 0; r < 16; ++r) {
      long e = e0 + r; long ee = (e < E) ? e : E - 1;
      long nd = ecen[ee];
      int col = lane; int m = col >> 1, jj = col & 1;
      float w0 = ws64[ee * 64 + 2 * m], w1 = ws64[ee * 64 + 2 * m + 1];
      const float* ef = eqfeat + ee * 4;
      float val;
      if (jj == 0) val = ef[0] * w0 * envT[nd * 64 + m * 4 + 1 + i];   // a_s*b_v
      else         val = ef[1 + i] * w1 * envT[nd * 64 + m * 4 + 0];   // a_v*b_s
      A[r * 32 + col] = (_Float16)val;
    }
    v8f vacc = splat8(0.0f);
    vacc = wmma16(a_frag(A, 32, 0), bfrag_ld(FB, FR_WLV), vacc);
#pragma unroll
    for (int j = 0; j < 8; ++j) {
      long e = e0 + j + hi * 8;
      if (e < E) out_eq[e * 64 + n * 4 + 1 + i] = vacc[j];
    }
  }
}

// ---------------------------------------------------------------- launch
extern "C" void kernel_launch(void* const* d_in, const int* in_sizes, int n_in,
                              void* d_out, int out_size, void* d_ws, size_t ws_size,
                              hipStream_t stream)
{
  (void)n_in; (void)out_size; (void)ws_size;
  const float* latents_in = (const float*)d_in[0];
  const float* X      = (const float*)d_in[1];
  const float* eqfeat = (const float*)d_in[2];
  const float* cutoff = (const float*)d_in[3];
  const float* eattr  = (const float*)d_in[4];
  const float* ninv   = (const float*)d_in[5];
  const float* einv   = (const float*)d_in[6];
  const float* W1   = (const float*)d_in[7];
  const float* b1   = (const float*)d_in[8];
  const float* W2   = (const float*)d_in[9];
  const float* b2   = (const float*)d_in[10];
  const float* Wenv = (const float*)d_in[11];
  const float* benv = (const float*)d_in[12];
  const float* lng  = (const float*)d_in[13];
  const float* lnb  = (const float*)d_in[14];
  const float* lnqg = (const float*)d_in[15];
  const float* lnqb = (const float*)d_in[16];
  const float* Wq   = (const float*)d_in[17];
  const float* bq   = (const float*)d_in[18];
  const float* Wd1  = (const float*)d_in[19];
  const float* Wd2  = (const float*)d_in[20];
  const float* lnkg = (const float*)d_in[21];
  const float* lnkb = (const float*)d_in[22];
  const float* Wk   = (const float*)d_in[23];
  const float* bk   = (const float*)d_in[24];
  const float* Ws   = (const float*)d_in[25];
  const float* Wv   = (const float*)d_in[26];
  const float* Wls  = (const float*)d_in[27];
  const float* Wlv  = (const float*)d_in[28];
  const int* active = (const int*)d_in[29];
  const int* ecen   = (const int*)d_in[30];
  const int* eneigh = (const int*)d_in[31];
  const int E = in_sizes[29];

  float* out_lat = (float*)d_out;
  float* out_inv = out_lat + (size_t)E * 128;
  float* out_eq  = out_inv + (size_t)E * 160;

  float* ws64  = (float*)d_ws;                          // [E,64] normalized env weights
  float* waBuf = ws64 + (size_t)E * 64;                 // [E,16] logits -> exp
  unsigned* mbuf = (unsigned*)(waBuf + (size_t)E * 16); // [NN,16] encoded seg max
  float* den  = (float*)(mbuf + (size_t)NN * 16);       // [NN,16] softmax denom
  float* env  = den + (size_t)NN * 16;                  // [NN,16,4] env sums
  float* envT = env + (size_t)NN * 64;                  // [NN,16,4] mixed env
  _Float16* FB = (_Float16*)(envT + (size_t)NN * 64);   // B-fragment table (208KB)

  // latents output starts as a copy of input (then scatter-overwritten)
  hipMemcpyAsync(out_lat, latents_in, (size_t)E * 128 * sizeof(float),
                 hipMemcpyDeviceToDevice, stream);

  repack_kernel<<<FR_TOTAL, 32, 0, stream>>>(W1, W2, Wenv, Wq, Wd1, Wd2,
                                             Wk, Ws, Wv, Wls, Wlv, FB);

  int nm = NN * 16, ne = NN * 64;
  init_ws_kernel<<<(ne + 255) / 256, 256, 0, stream>>>(mbuf, den, env, nm, ne);

  int tiles = (E + 15) / 16;
  int blocksA = (tiles + 1) / 2;
  edge_main_kernel<<<blocksA, 64, 0, stream>>>(
      X, eqfeat, cutoff, eattr, ninv, einv,
      b1, b2, benv, lng, lnb, lnqg, lnqb, bq, lnkg, lnkb, bk,
      active, ecen, eneigh, FB, out_lat, ws64, waBuf, mbuf, E);

  int n16 = E * 16;
  softmax_kernel<<<(n16 + 255) / 256, 256, 0, stream>>>(waBuf, ecen, mbuf, den, n16);
  env_accum_kernel<<<(n16 + 255) / 256, 256, 0, stream>>>(waBuf, den, ws64, eattr, ecen, env, n16);

  int ntiles = (NN + 15) / 16;
  env_xform_kernel<<<(ntiles + 3) / 4, 128, 0, stream>>>(env, FB, envT, NN);

  edge_out_kernel<<<blocksA, 128, 0, stream>>>(
      eqfeat, ws64, envT, FB, active, ecen, out_lat, out_inv, out_eq, E);
}
